// SMNet_15255723835828
// MI455X (gfx1250) — matrix operand order repaired and checked by reading, not compiled, not run on hardware
//
#include <hip/hip_runtime.h>
#include <hip/hip_bf16.h>
#include <stdint.h>

// ---------------- types for CDNA5 WMMA ----------------
typedef __bf16 v16bf __attribute__((ext_vector_type(16)));
typedef float  v8f   __attribute__((ext_vector_type(8)));
typedef unsigned int u32x4 __attribute__((ext_vector_type(4)));

union FragAB { v16bf v; u32x4 q[2]; };

// ---------------- problem constants ----------------
constexpr int MHh = 250, MWw = 250, MCELL = MHh * MWw;   // 62500 map cells
constexpr int MTILES = (MCELL + 15) / 16;                // 3907
constexpr int MPAD = MTILES * 16;                        // 62512
constexpr int Tf = 10, CF = 64, IH = 480, IW = 640;
constexpr int Pp = IH * IW;                              // 307200 pixels / frame
constexpr int DH = 256, G3 = 768;                        // hidden, 3*hidden
constexpr int SP = 256;                                  // padded spatial dim (250 + 2*3 -> round to 256)

#define DEVI static __device__ __forceinline__

DEVI unsigned short f2bf(float f) {
    union { float f; unsigned int u; } x; x.f = f;
    unsigned int r = x.u + 0x7FFFu + ((x.u >> 16) & 1u);   // RNE
    return (unsigned short)(r >> 16);
}

// A-matrix 16x32 bf16 fragment (ISA 7.12.2 interleaved layout):
// lane row = lane&15 (caller bakes into base), hi = lane>>4 selects K-halves.
DEVI v16bf load_a(const unsigned short* rowbase, int kbase, int hi) {
    FragAB f;
    f.q[0] = *(const u32x4*)(rowbase + kbase + hi * 8);        // K pairs k..k+7
    f.q[1] = *(const u32x4*)(rowbase + kbase + 16 + hi * 8);   // K pairs k+16..k+23
    return f.v;
}
// B-matrix 32x16 bf16 fragment: lane col = lane&15 (baked into wrow), hi selects K-halves,
// same interleaved K order as load_a.
DEVI v16bf load_b(const unsigned short* wrow, int kbase, int hi) {
    FragAB f;
    f.q[0] = *(const u32x4*)(wrow + kbase + hi * 8);
    f.q[1] = *(const u32x4*)(wrow + kbase + 16 + hi * 8);
    return f.v;
}

DEVI float sigmoidf(float x) { return 1.0f / (1.0f + __expf(-x)); }

// ---------------- utility kernels ----------------
__global__ void k_zero(unsigned int* p, size_t n) {
    size_t i = (size_t)blockIdx.x * blockDim.x + threadIdx.x;
    size_t st = (size_t)gridDim.x * blockDim.x;
    for (; i < n; i += st) p[i] = 0u;
}

__global__ void k_f2bf(const float* __restrict__ s, unsigned short* __restrict__ d, int n) {
    int i = blockIdx.x * blockDim.x + threadIdx.x;
    if (i < n) d[i] = f2bf(s[i]);
}

// OIHW f32 -> [Op][kh][kw][Ip] bf16 (channel-padded, row-padded, zeros elsewhere)
__global__ void k_conv_w(const float* __restrict__ w, unsigned short* __restrict__ wt,
                         int O, int I, int Ip, int K, int Op) {
    int total = Op * K * K * Ip;
    int i = blockIdx.x * blockDim.x + threadIdx.x;
    if (i >= total) return;
    int ci = i % Ip; int r = i / Ip;
    int kw = r % K;  r /= K;
    int kh = r % K;  int o = r / K;
    float v = 0.0f;
    if (o < O && ci < I) v = w[(((size_t)o * I + ci) * K + kh) * K + kw];
    wt[((size_t)o * K * K + kh * K + kw) * Ip + ci] = f2bf(v);
}

// ---------------- per-frame kernels ----------------
__global__ void k_prep(const float* __restrict__ hmap, float* __restrict__ hold,
                       int* __restrict__ winner) {
    int i = blockIdx.x * blockDim.x + threadIdx.x;
    if (i < MCELL) { hold[i] = hmap[i]; winner[i] = -1; }
}

__global__ void k_scatter(const int* __restrict__ proj, const unsigned char* __restrict__ mask,
                          const float* __restrict__ hts, float* __restrict__ hmap) {
    int p = blockIdx.x * blockDim.x + threadIdx.x;
    if (p >= Pp) return;
    if (mask[p]) return;                                  // outlier: -inf, never wins
    int idx = MWw * proj[2 * p + 1] + proj[2 * p];
    float h = hts[p] + 1000.0f;                           // positive -> int-compare == float-compare
    atomicMax((int*)hmap + idx, __float_as_int(h));
}

__global__ void k_winner(const int* __restrict__ proj, const unsigned char* __restrict__ mask,
                         const float* __restrict__ hts, const float* __restrict__ hmap,
                         int* __restrict__ winner) {
    int p = blockIdx.x * blockDim.x + threadIdx.x;
    if (p >= Pp) return;
    if (mask[p]) return;
    int idx = MWw * proj[2 * p + 1] + proj[2 * p];
    float h = hts[p] + 1000.0f;
    if (h == hmap[idx]) atomicMax(&winner[idx], p);
}

__global__ void k_gather(const float* __restrict__ feat, const int* __restrict__ winner,
                         const float* __restrict__ hmap, const float* __restrict__ hold,
                         int* __restrict__ upd, int* __restrict__ obs,
                         unsigned short* __restrict__ xb) {
    int i = blockIdx.x * blockDim.x + threadIdx.x;
    if (i >= MCELL * CF) return;
    int m = i >> 6, c = i & 63;
    int w = winner[m]; if (w < 0) w = 0;
    xb[(size_t)m * CF + c] = f2bf(feat[(size_t)c * Pp + w]);
    if (c == 0) {
        int u = hmap[m] > hold[m] ? 1 : 0;
        upd[m] = u;
        if (u) obs[m] = 1;
    }
}

// ---------------- fused GRU step (WMMA) ----------------
// block = 16 rows of M, 8 waves; wave w owns d-tiles {2w, 2w+1} (16 cols each),
// and for each d-tile accumulates 3 gate tiles x {gi, gh} entirely in registers.
__global__ void k_gru(const unsigned short* __restrict__ xb,    // [MPAD][64]  bf16
                      unsigned short* __restrict__ hb,          // [MPAD][256] bf16 (RW)
                      const unsigned short* __restrict__ wih,   // [768][64]   bf16
                      const unsigned short* __restrict__ whh,   // [768][256]  bf16
                      const float* __restrict__ bih, const float* __restrict__ bhh,
                      float* __restrict__ state,                // [MPAD][256] f32 (RW)
                      const int* __restrict__ upd) {
    const int lane = threadIdx.x & 31;
    const int wv   = threadIdx.x >> 5;
    const int nl   = lane & 15;
    const int hi   = lane >> 4;
    const int m0   = blockIdx.x * 16;

    v8f ai[2][3], ah[2][3];
    #pragma unroll
    for (int a = 0; a < 2; ++a)
        #pragma unroll
        for (int g = 0; g < 3; ++g)
            #pragma unroll
            for (int v = 0; v < 8; ++v) { ai[a][g][v] = 0.0f; ah[a][g][v] = 0.0f; }

    const unsigned short* xrow = xb + (size_t)(m0 + nl) * CF;
    const unsigned short* hrow = hb + (size_t)(m0 + nl) * DH;

    // Gi = X @ Wih^T : K = 64
    #pragma unroll
    for (int ks = 0; ks < 2; ++ks) {
        v16bf a = load_a(xrow, ks * 32, hi);
        #pragma unroll
        for (int dt = 0; dt < 2; ++dt)
            #pragma unroll
            for (int g = 0; g < 3; ++g) {
                int n = g * 256 + (wv * 2 + dt) * 16 + nl;
                v16bf b = load_b(wih + (size_t)n * CF, ks * 32, hi);
                ai[dt][g] = __builtin_amdgcn_wmma_f32_16x16x32_bf16(
                    false, a, false, b, (short)0, ai[dt][g], false, false);
            }
    }
    // Gh = H @ Whh^T : K = 256
    for (int ks = 0; ks < 8; ++ks) {
        v16bf a = load_a(hrow, ks * 32, hi);
        #pragma unroll
        for (int dt = 0; dt < 2; ++dt)
            #pragma unroll
            for (int g = 0; g < 3; ++g) {
                int n = g * 256 + (wv * 2 + dt) * 16 + nl;
                v16bf b = load_b(whh + (size_t)n * DH, ks * 32, hi);
                ah[dt][g] = __builtin_amdgcn_wmma_f32_16x16x32_bf16(
                    false, a, false, b, (short)0, ah[dt][g], false, false);
            }
    }

    // elementwise GRU update, fully in registers (gates r,z,n at d, d+256, d+512)
    #pragma unroll
    for (int dt = 0; dt < 2; ++dt) {
        int d = (wv * 2 + dt) * 16 + nl;
        float bri = bih[d],       brh = bhh[d];
        float bzi = bih[256 + d], bzh = bhh[256 + d];
        float bni = bih[512 + d], bnh = bhh[512 + d];
        #pragma unroll
        for (int v = 0; v < 8; ++v) {
            int m = m0 + v + 8 * hi;
            if (m < MCELL && upd[m]) {
                float r  = sigmoidf(ai[dt][0][v] + ah[dt][0][v] + bri + brh);
                float z  = sigmoidf(ai[dt][1][v] + ah[dt][1][v] + bzi + bzh);
                float nn = tanhf(ai[dt][2][v] + bni + r * (ah[dt][2][v] + bnh));
                size_t o = (size_t)m * DH + d;
                float hnew = (1.0f - z) * nn + z * state[o];
                state[o] = hnew;
                hb[o] = f2bf(hnew);
            }
        }
    }
}

// ---------------- conv stage (implicit-GEMM WMMA) ----------------
__global__ void k_fillact(const float* __restrict__ state, unsigned short* __restrict__ act) {
    int i = blockIdx.x * blockDim.x + threadIdx.x;
    if (i >= MCELL * DH) return;
    int m = i >> 8, c = i & 255;
    int y = m / MWw, x = m % MWw;
    act[((size_t)((y + 3) * SP + x + 3)) * DH + c] = f2bf(state[(size_t)m * DH + c]);
}

// act: channel-last bf16 [SP][SP][Cin], zero borders; wt: [n][kh][kw][Cin] bf16.
// Each wave owns 2 N-tiles (independent WMMA chains).
template <int KS, int PADQ, bool FINAL>
__global__ void k_conv(const unsigned short* __restrict__ act, int Cin,
                       const unsigned short* __restrict__ wt,
                       float* __restrict__ outp, int Coutp, int Ntiles,
                       const float* __restrict__ bias, float* __restrict__ dout, int CoutReal) {
    const int lane = threadIdx.x & 31;
    const int wv   = threadIdx.x >> 5;
    const int nl   = lane & 15;
    const int hi   = lane >> 4;
    const int m0   = blockIdx.x * 16;

    int mm = m0 + nl; if (mm >= MCELL) mm = MCELL - 1;   // clamp tail tile (masked at write)
    const int y = mm / MWw, x = mm % MWw;

    const int nt0 = wv * 2, nt1 = wv * 2 + 1;
    const bool has1 = (nt1 < Ntiles);                    // wave-uniform
    const int rowLen = KS * KS * Cin;
    const unsigned short* w0 = wt + (size_t)(nt0 * 16 + nl) * rowLen;
    const unsigned short* w1 = wt + (size_t)((has1 ? nt1 : nt0) * 16 + nl) * rowLen;

    v8f acc0, acc1;
    #pragma unroll
    for (int v = 0; v < 8; ++v) { acc0[v] = 0.0f; acc1[v] = 0.0f; }

    for (int kh = 0; kh < KS; ++kh) {
        const int py = y + kh + (3 - PADQ);
        for (int kw = 0; kw < KS; ++kw) {
            const int px = x + kw + (3 - PADQ);
            const unsigned short* abase = act + ((size_t)(py * SP + px)) * Cin;
            const unsigned short* wb0 = w0 + (kh * KS + kw) * Cin;
            const unsigned short* wb1 = w1 + (kh * KS + kw) * Cin;
            for (int cs = 0; cs < Cin; cs += 32) {
                v16bf a = load_a(abase, cs, hi);
                v16bf b0 = load_b(wb0, cs, hi);
                acc0 = __builtin_amdgcn_wmma_f32_16x16x32_bf16(
                    false, a, false, b0, (short)0, acc0, false, false);
                if (has1) {
                    v16bf b1 = load_b(wb1, cs, hi);
                    acc1 = __builtin_amdgcn_wmma_f32_16x16x32_bf16(
                        false, a, false, b1, (short)0, acc1, false, false);
                }
            }
        }
    }

    #pragma unroll
    for (int v = 0; v < 8; ++v) {
        int m = m0 + v + 8 * hi;
        if (m >= MCELL) continue;
        int n0 = nt0 * 16 + nl;
        if (FINAL) {
            if (n0 < CoutReal) dout[(size_t)n0 * MCELL + m] = acc0[v] + bias[n0];
        } else {
            int yy = m / MWw, xx = m % MWw;
            outp[((size_t)((yy + 3) * SP + xx + 3)) * Coutp + n0] = acc0[v];
        }
        if (has1) {
            int n1 = nt1 * 16 + nl;
            if (FINAL) {
                if (n1 < CoutReal) dout[(size_t)n1 * MCELL + m] = acc1[v] + bias[n1];
            } else {
                int yy = m / MWw, xx = m % MWw;
                outp[((size_t)((yy + 3) * SP + xx + 3)) * Coutp + n1] = acc1[v];
            }
        }
    }
}

// training-mode BN: population mean/var over the 62500 valid pixels, per channel
__global__ void k_bnstats(const float* __restrict__ outp, int Coutp, float* __restrict__ stats) {
    __shared__ float s1[256], s2[256];
    const int c = blockIdx.x, tid = threadIdx.x;
    float sum = 0.0f, sq = 0.0f;
    for (int i = tid; i < MCELL; i += 256) {
        int y = i / MWw, x = i % MWw;
        float v = outp[((size_t)((y + 3) * SP + x + 3)) * Coutp + c];
        sum += v; sq += v * v;
    }
    s1[tid] = sum; s2[tid] = sq; __syncthreads();
    for (int s = 128; s > 0; s >>= 1) {
        if (tid < s) { s1[tid] += s1[tid + s]; s2[tid] += s2[tid + s]; }
        __syncthreads();
    }
    if (tid == 0) {
        float mu = s1[0] / (float)MCELL;
        float var = s2[0] / (float)MCELL - mu * mu;
        stats[2 * c] = mu;
        stats[2 * c + 1] = rsqrtf(var + 1e-5f);
    }
}

__global__ void k_bnapply(const float* __restrict__ outp, int Coutp, int Cout,
                          const float* __restrict__ stats,
                          const float* __restrict__ g, const float* __restrict__ b,
                          unsigned short* __restrict__ actn, int Cnp) {
    int i = blockIdx.x * blockDim.x + threadIdx.x;
    if (i >= MCELL * Cout) return;
    int c = i % Cout, m = i / Cout;
    int y = m / MWw, x = m % MWw;
    size_t pix = (size_t)((y + 3) * SP + x + 3);
    float v = outp[pix * Coutp + c];
    float o = (v - stats[2 * c]) * stats[2 * c + 1] * g[c] + b[c];
    o = o > 0.0f ? o : 0.0f;
    actn[pix * Cnp + c] = f2bf(o);
}

__global__ void k_outputs(const int* __restrict__ obs, const float* __restrict__ hmap,
                          float* __restrict__ dout) {
    int i = blockIdx.x * blockDim.x + threadIdx.x;
    if (i < MCELL) {
        dout[20 * MCELL + i] = obs[i] ? 1.0f : 0.0f;   // observed map
        dout[21 * MCELL + i] = hmap[i];                // height map (carries +1000 offset, as ref)
    }
}

// ---------------- host orchestration ----------------
extern "C" void kernel_launch(void* const* d_in, const int* in_sizes, int n_in,
                              void* d_out, int out_size, void* d_ws, size_t ws_size,
                              hipStream_t stream) {
    (void)in_sizes; (void)n_in; (void)out_size; (void)ws_size;

    const float*         features = (const float*)d_in[0];
    const int*           proj     = (const int*)d_in[1];
    const unsigned char* mask     = (const unsigned char*)d_in[2];
    const float*         heights  = (const float*)d_in[3];
    const float* w_ih = (const float*)d_in[6];
    const float* w_hh = (const float*)d_in[7];
    const float* b_ih = (const float*)d_in[8];
    const float* b_hh = (const float*)d_in[9];
    const float* c1w = (const float*)d_in[10];
    const float* g1 = (const float*)d_in[11]; const float* bb1 = (const float*)d_in[12];
    const float* c2w = (const float*)d_in[13];
    const float* g2 = (const float*)d_in[14]; const float* bb2 = (const float*)d_in[15];
    const float* c3w = (const float*)d_in[16];
    const float* g3 = (const float*)d_in[17]; const float* bb3 = (const float*)d_in[18];
    const float* c4w = (const float*)d_in[19];
    const float* g4 = (const float*)d_in[20]; const float* bb4 = (const float*)d_in[21];
    const float* c5w = (const float*)d_in[22];
    const float* c5b = (const float*)d_in[23];

    // ---- workspace layout ----
    char* base = (char*)d_ws;
    size_t off = 0;
    auto alloc = [&](size_t bytes) -> void* {
        void* p = base + off;
        off = (off + bytes + 255) & ~(size_t)255;
        return p;
    };
    float* hmap   = (float*)alloc((size_t)MCELL * 4);
    float* hold   = (float*)alloc((size_t)MCELL * 4);
    int*   winner = (int*)  alloc((size_t)MCELL * 4);
    int*   upd    = (int*)  alloc((size_t)MCELL * 4);
    int*   obs    = (int*)  alloc((size_t)MCELL * 4);
    float* state  = (float*)alloc((size_t)MPAD * DH * 4);
    unsigned short* hb   = (unsigned short*)alloc((size_t)MPAD * DH * 2);
    unsigned short* xb   = (unsigned short*)alloc((size_t)MPAD * CF * 2);
    unsigned short* wihb = (unsigned short*)alloc((size_t)G3 * CF * 2);
    unsigned short* whhb = (unsigned short*)alloc((size_t)G3 * DH * 2);
    unsigned short* w1t  = (unsigned short*)alloc((size_t)128 * 49 * 256 * 2);
    unsigned short* w2t  = (unsigned short*)alloc((size_t)64 * 9 * 128 * 2);
    unsigned short* w3t  = (unsigned short*)alloc((size_t)48 * 9 * 64 * 2);
    unsigned short* w4t  = (unsigned short*)alloc((size_t)48 * 9 * 64 * 2);
    unsigned short* w5t  = (unsigned short*)alloc((size_t)32 * 64 * 2);
    unsigned short* act0 = (unsigned short*)alloc((size_t)SP * SP * 256 * 2);
    unsigned short* act1 = (unsigned short*)alloc((size_t)SP * SP * 128 * 2);
    float* convout = (float*)alloc((size_t)SP * SP * 128 * 4);
    float* stats   = (float*)alloc((size_t)128 * 2 * 4);

    auto zero = [&](void* p, size_t dwords) {
        size_t blocks = (dwords + 255) / 256;
        if (blocks > 4096) blocks = 4096;
        k_zero<<<dim3((unsigned)blocks), dim3(256), 0, stream>>>((unsigned int*)p, dwords);
    };

    // ---- deterministic init (harness does not re-poison) ----
    zero(hmap, MCELL);
    zero(obs, MCELL);
    zero(state, (size_t)MPAD * DH);
    zero(hb, (size_t)MPAD * DH / 2);
    zero(xb, (size_t)MPAD * CF / 2);

    // ---- weight preprocessing (f32 -> bf16, conv weights to [O][kh][kw][Ci_pad]) ----
    k_f2bf<<<(G3 * CF + 255) / 256, 256, 0, stream>>>(w_ih, wihb, G3 * CF);
    k_f2bf<<<(G3 * DH + 255) / 256, 256, 0, stream>>>(w_hh, whhb, G3 * DH);
    k_conv_w<<<(128 * 49 * 256 + 255) / 256, 256, 0, stream>>>(c1w, w1t, 128, 256, 256, 7, 128);
    k_conv_w<<<(64 * 9 * 128 + 255) / 256, 256, 0, stream>>>(c2w, w2t, 64, 128, 128, 3, 64);
    k_conv_w<<<(48 * 9 * 64 + 255) / 256, 256, 0, stream>>>(c3w, w3t, 48, 64, 64, 3, 48);
    k_conv_w<<<(48 * 9 * 64 + 255) / 256, 256, 0, stream>>>(c4w, w4t, 48, 48, 64, 3, 48);
    k_conv_w<<<(32 * 64 + 255) / 256, 256, 0, stream>>>(c5w, w5t, 20, 48, 64, 1, 32);

    // ---- temporal scan: scatter-max + winner + gather + fused WMMA GRU ----
    for (int t = 0; t < Tf; ++t) {
        const int* projT = proj + (size_t)t * Pp * 2;
        const unsigned char* maskT = mask + (size_t)t * Pp;
        const float* htsT = heights + (size_t)t * Pp;
        const float* featT = features + (size_t)t * CF * Pp;

        k_prep<<<(MCELL + 255) / 256, 256, 0, stream>>>(hmap, hold, winner);
        k_scatter<<<(Pp + 255) / 256, 256, 0, stream>>>(projT, maskT, htsT, hmap);
        k_winner<<<(Pp + 255) / 256, 256, 0, stream>>>(projT, maskT, htsT, hmap, winner);
        k_gather<<<(MCELL * CF + 255) / 256, 256, 0, stream>>>(featT, winner, hmap, hold,
                                                               upd, obs, xb);
        k_gru<<<MTILES, 256, 0, stream>>>(xb, hb, wihb, whhb, b_ih, b_hh, state, upd);
    }

    // ---- conv head ----
    // memory -> act0 (channel-last, pad 3, bf16)
    zero(act0, (size_t)SP * SP * 256 / 2);
    k_fillact<<<(MCELL * DH + 255) / 256, 256, 0, stream>>>(state, act0);

    // conv1 7x7 256->128, BN, ReLU
    zero(convout, (size_t)SP * SP * 128);
    k_conv<7, 3, false><<<MTILES, 128, 0, stream>>>(act0, 256, w1t, convout, 128, 8,
                                                    nullptr, nullptr, 0);
    k_bnstats<<<128, 256, 0, stream>>>(convout, 128, stats);
    zero(act1, (size_t)SP * SP * 128 / 2);
    k_bnapply<<<(MCELL * 128 + 255) / 256, 256, 0, stream>>>(convout, 128, 128, stats, g1, bb1,
                                                             act1, 128);
    // conv2 3x3 128->64
    zero(convout, (size_t)SP * SP * 64);
    k_conv<3, 1, false><<<MTILES, 64, 0, stream>>>(act1, 128, w2t, convout, 64, 4,
                                                   nullptr, nullptr, 0);
    k_bnstats<<<64, 256, 0, stream>>>(convout, 64, stats);
    zero(act0, (size_t)SP * SP * 64 / 2);
    k_bnapply<<<(MCELL * 64 + 255) / 256, 256, 0, stream>>>(convout, 64, 64, stats, g2, bb2,
                                                            act0, 64);
    // conv3 3x3 64->48 (next act channel-padded to 64)
    zero(convout, (size_t)SP * SP * 48);
    k_conv<3, 1, false><<<MTILES, 64, 0, stream>>>(act0, 64, w3t, convout, 48, 3,
                                                   nullptr, nullptr, 0);
    k_bnstats<<<48, 256, 0, stream>>>(convout, 48, stats);
    zero(act1, (size_t)SP * SP * 64 / 2);
    k_bnapply<<<(MCELL * 48 + 255) / 256, 256, 0, stream>>>(convout, 48, 48, stats, g3, bb3,
                                                            act1, 64);
    // conv4 3x3 48->48
    zero(convout, (size_t)SP * SP * 48);
    k_conv<3, 1, false><<<MTILES, 64, 0, stream>>>(act1, 64, w4t, convout, 48, 3,
                                                   nullptr, nullptr, 0);
    k_bnstats<<<48, 256, 0, stream>>>(convout, 48, stats);
    zero(act0, (size_t)SP * SP * 64 / 2);
    k_bnapply<<<(MCELL * 48 + 255) / 256, 256, 0, stream>>>(convout, 48, 48, stats, g4, bb4,
                                                            act0, 64);
    // conv5 1x1 48->20 + bias, straight into d_out (NCHW dense)
    k_conv<1, 0, true><<<MTILES, 32, 0, stream>>>(act0, 64, w5t, nullptr, 0, 2,
                                                  c5b, (float*)d_out, 20);

    // observed + hmap outputs
    k_outputs<<<(MCELL + 255) / 256, 256, 0, stream>>>(obs, hmap, (float*)d_out);
}
